// DIN_55173149884770
// MI455X (gfx1250) — matrix operand order repaired and checked by reading, not compiled, not run on hardware
//
#include <hip/hip_runtime.h>
#include <math.h>

// ---------------------------------------------------------------------------
// DIN forward for MI455X (gfx1250, wave32, WMMA 16x16x32 f16->f32)
// ---------------------------------------------------------------------------

#define Bn   2048
#define Tn   100
#define TP   128          // padded sequence rows (8 tiles of 16)
#define Dn   64
#define DKn  128
#define OTHER_V 10000
#define A1n  80           // attention hidden 1 (5 n-tiles)
#define A1P  96           // padded as K for layer 2 (3 k-steps)
#define A2n  40
#define A2P  48           // padded N for layer 2 (3 n-tiles)
#define FINn 388
#define FINP 416          // 13 k-steps of 32
#define NEGV (-4294967295.0f)
#define INV_SQRT_DK 0.08838834764831845f

typedef __attribute__((ext_vector_type(16))) _Float16 v16h;
typedef __attribute__((ext_vector_type(8)))  _Float16 h8;
typedef __attribute__((ext_vector_type(8)))  float    v8f;

__device__ __forceinline__ v8f wmma16(v16h a, v16h b, v8f c) {
    return __builtin_amdgcn_wmma_f32_16x16x32_f16(
        /*neg_a=*/false, a, /*neg_b=*/false, b,
        /*c_mod=*/(short)0, c, /*reuse_a=*/false, /*reuse_b=*/false);
}

// A-fragment (16x32 f16) from an LDS tile with row pitch `pitch` (f16 units).
// Lane L: row = L&15; lane-half selects K sub-blocks {8*hi..} and {16+8*hi..}.
__device__ __forceinline__ v16h ldsA_frag(const _Float16* base, int pitch,
                                          int kt, int lane) {
    int r = lane & 15, hi = lane >> 4;
    const _Float16* row = base + r * pitch + kt * 32 + 8 * hi;
    union { v16h v; h8 h[2]; } u;
    u.h[0] = *(const h8*)(row);
    u.h[1] = *(const h8*)(row + 16);
    return u.v;
}

// B-fragment (32x16 f16) from a transposed weight matrix Wt[N][K] in global.
// Lane L: column N = nt*16 + (L&15); lane-half selects K sub-blocks {8*hi..} and {16+8*hi..}.
__device__ __forceinline__ v16h glbB_frag(const _Float16* Wt, int pitch,
                                          int nt, int kt, int lane) {
    int c = lane & 15, hi = lane >> 4;
    const _Float16* p = Wt + (size_t)(nt * 16 + c) * pitch + kt * 32 + hi * 8;
    union { v16h v; h8 h[2]; } u;
    u.h[0] = *(const h8*)(p);
    u.h[1] = *(const h8*)(p + 16);
    return u.v;
}

// D-fragment (16x16 f32) -> bias + PReLU -> f16 LDS tile (row pitch `pitch`)
__device__ __forceinline__ void store_act_f16(_Float16* dst, int pitch, int nt,
                                              v8f acc, const float* bias,
                                              const float* alpha, int lane,
                                              int nValid) {
    int c = lane & 15, hi = lane >> 4;
    int n = nt * 16 + c;
    float bv = (n < nValid) ? bias[n] : 0.f;
    float av = (n < nValid) ? alpha[n] : 0.f;
#pragma unroll
    for (int v = 0; v < 8; ++v) {
        int row = v + 8 * hi;
        float x = acc[v] + bv;
        x = x > 0.f ? x : av * x;
        dst[row * pitch + n] = (_Float16)x;
    }
}

__device__ __forceinline__ void store_act_f32(float* dst, int pitch, int nt,
                                              v8f acc, const float* bias,
                                              const float* alpha, int lane,
                                              int nValid) {
    int c = lane & 15, hi = lane >> 4;
    int n = nt * 16 + c;
    float bv = (n < nValid) ? bias[n] : 0.f;
    float av = (n < nValid) ? alpha[n] : 0.f;
#pragma unroll
    for (int v = 0; v < 8; ++v) {
        int row = v + 8 * hi;
        float x = acc[v] + bv;
        x = x > 0.f ? x : av * x;
        dst[row * pitch + n] = x;
    }
}

// info = [q, k, q-k, q*k]; K region = k>>7; chunks of 8 never straddle regions.
__device__ __forceinline__ h8 info_chunk(const _Float16* q, const _Float16* s, int k) {
    int off = k & 127;
    h8 qv = *(const h8*)(q + off);
    h8 sv = *(const h8*)(s + off);
    switch (k >> 7) {
        case 0:  return qv;
        case 1:  return sv;
        case 2:  return qv - sv;
        default: return qv * sv;
    }
}

__device__ __forceinline__ v16h info_frag(const _Float16* q, const _Float16* seqTile,
                                          int kt, int lane) {
    int r = lane & 15, hi = lane >> 4;
    const _Float16* s = seqTile + r * DKn;
    int k0 = kt * 32 + 8 * hi;
    union { v16h v; h8 h[2]; } u;
    u.h[0] = info_chunk(q, s, k0);
    u.h[1] = info_chunk(q, s, k0 + 16);
    return u.v;
}

// ---------------------------------------------------------------------------
// Kernel 0: f32 [K,N] -> f16 transposed + zero-padded [Np,Kp]
// ---------------------------------------------------------------------------
__global__ void transpose_pad_f16(const float* __restrict__ src,
                                  _Float16* __restrict__ dst,
                                  int K, int N, int Kp, int Np) {
    int i = blockIdx.x * blockDim.x + threadIdx.x;
    if (i >= Np * Kp) return;
    int n = i / Kp, k = i % Kp;
    float v = (n < N && k < K) ? src[k * N + n] : 0.f;
    dst[i] = (_Float16)v;
}

// ---------------------------------------------------------------------------
// Kernel 1: attention + softmax + pooling. 1 block (4 waves) per batch row.
// ---------------------------------------------------------------------------
__global__ void __launch_bounds__(128)
din_attention(const int* __restrict__ seq_inputs,     // [B,T,2]
              const int* __restrict__ item_inputs,    // [B,2]
              const float* __restrict__ embed_item,   // [ITEM_V,64]
              const float* __restrict__ embed_cate,   // [ITEM_V,64]
              const _Float16* __restrict__ W1t,       // [80,512]
              const _Float16* __restrict__ W2t,       // [48,96]
              const float* __restrict__ att_b1, const float* __restrict__ att_a1,
              const float* __restrict__ att_b2, const float* __restrict__ att_a2,
              const float* __restrict__ att_W3, const float* __restrict__ att_b3,
              float* __restrict__ user_ws,            // [B,128] f32
              _Float16* __restrict__ qitem_ws) {      // [B,128] f16
    __shared__ _Float16 qf16[DKn];
    __shared__ _Float16 seqf16[TP][DKn];        // 32 KB
    __shared__ _Float16 h1buf[4][2][16 * A1P];  // 24 KB
    __shared__ float    h2buf[4][16 * A2P];     // 12 KB
    __shared__ float    scores[TP];
    __shared__ float    wts[TP];
    __shared__ float    maskL[TP];
    __shared__ float    red[2];

    const int b    = blockIdx.x;
    const int tid  = threadIdx.x;
    const int lane = tid & 31;
    const int wv   = tid >> 5;

    // ---- gather target item (q) ----
    int it0 = item_inputs[b * 2 + 0];
    int it1 = item_inputs[b * 2 + 1];
    if (tid < DKn) {
        float v = (tid < Dn) ? embed_item[(size_t)it0 * Dn + tid]
                             : embed_cate[(size_t)it1 * Dn + (tid - Dn)];
        qf16[tid] = (_Float16)v;
        qitem_ws[(size_t)b * DKn + tid] = (_Float16)v;
    }
    // zero padded h1 columns (80..95) once
    for (int e = tid; e < 4 * 2 * 16 * A1P; e += 128)
        ((_Float16*)h1buf)[e] = (_Float16)0.f;

    // ---- gather behavior sequence rows (wave w takes rows w, w+4, ...) ----
    for (int t = wv; t < TP; t += 4) {
        int id0 = 0, id1 = 0;
        if (t < Tn) {
            id0 = seq_inputs[((size_t)b * Tn + t) * 2 + 0];
            id1 = seq_inputs[((size_t)b * Tn + t) * 2 + 1];
        }
        if (lane == 0) maskL[t] = (t < Tn && id0 != 0) ? 1.f : 0.f;
        for (int c = lane; c < DKn; c += 32) {
            float v = 0.f;
            if (t < Tn)
                v = (c < Dn) ? embed_item[(size_t)id0 * Dn + c]
                             : embed_cate[(size_t)id1 * Dn + (c - Dn)];
            seqf16[t][c] = (_Float16)v;
        }
    }
    __syncthreads();

    // ---- attention layer 1: [16x512] x [512x80], two T-tiles per wave ----
    const int t0 = 2 * wv, t1 = 2 * wv + 1;
    const _Float16* s0 = &seqf16[t0 * 16][0];
    const _Float16* s1 = &seqf16[t1 * 16][0];
    for (int nt = 0; nt < 5; ++nt) {
        v8f acc0 = {}, acc1 = {};
        for (int kt = 0; kt < 16; ++kt) {
            v16h bf = glbB_frag(W1t, 512, nt, kt, lane);
            v16h a0 = info_frag(qf16, s0, kt, lane);
            v16h a1 = info_frag(qf16, s1, kt, lane);
            acc0 = wmma16(a0, bf, acc0);
            acc1 = wmma16(a1, bf, acc1);
        }
        store_act_f16(&h1buf[wv][0][0], A1P, nt, acc0, att_b1, att_a1, lane, A1n);
        store_act_f16(&h1buf[wv][1][0], A1P, nt, acc1, att_b1, att_a1, lane, A1n);
    }
    __syncthreads();

    // ---- attention layers 2 + 3 ----
    const float w3b = att_b3[0];
    for (int tt = 0; tt < 2; ++tt) {
        for (int nt = 0; nt < 3; ++nt) {
            v8f acc = {};
            for (int kt = 0; kt < 3; ++kt) {
                v16h a  = ldsA_frag(&h1buf[wv][tt][0], A1P, kt, lane);
                v16h bf = glbB_frag(W2t, A1P, nt, kt, lane);
                acc = wmma16(a, bf, acc);
            }
            store_act_f32(&h2buf[wv][0], A2P, nt, acc, att_b2, att_a2, lane, A2n);
        }
        __syncthreads();
        if (lane < 16) {
            int row = (2 * wv + tt) * 16 + lane;
            float s = w3b;
            for (int n = 0; n < A2n; ++n)
                s += h2buf[wv][lane * A2P + n] * att_W3[n];
            s = (maskL[row] != 0.f) ? s : NEGV;
            scores[row] = s * INV_SQRT_DK;
        }
        __syncthreads();
    }

    // ---- masked softmax over T ----
    if (tid == 0) {
        float m = -3.4e38f;
        for (int t = 0; t < Tn; ++t) m = fmaxf(m, scores[t]);
        red[0] = m;
    }
    __syncthreads();
    float mx = red[0];
    wts[tid] = (tid < Tn) ? __expf(scores[tid] - mx) : 0.f;
    __syncthreads();
    if (tid == 0) {
        float s = 0.f;
        for (int t = 0; t < Tn; ++t) s += wts[t];
        red[1] = s;
    }
    __syncthreads();
    float inv = 1.f / red[1];

    // ---- weighted-sum pooling: user_info[d] = sum_t w[t]*seq[t][d] ----
    {
        float acc = 0.f;
        for (int t = 0; t < Tn; ++t)
            acc += wts[t] * (float)seqf16[t][tid];
        user_ws[(size_t)b * DKn + tid] = acc * inv;
    }
}

// ---------------------------------------------------------------------------
// Kernel 2: FFN 388->80->40->1 + sigmoid. 1 wave per 16 batch rows.
// ---------------------------------------------------------------------------
__global__ void __launch_bounds__(32)
din_ffn(const float* __restrict__ dense_inputs,   // [B,4]
        const int* __restrict__ sparse_inputs,    // [B,2]
        const float* __restrict__ embed_other,    // [2,OTHER_V,64]
        const float* __restrict__ user_ws,        // [B,128]
        const _Float16* __restrict__ qitem_ws,    // [B,128]
        const _Float16* __restrict__ F1t,         // [80,416]
        const _Float16* __restrict__ F2t,         // [48,96]
        const float* __restrict__ ffn_b1, const float* __restrict__ ffn_a1,
        const float* __restrict__ ffn_b2, const float* __restrict__ ffn_a2,
        const float* __restrict__ out_W, const float* __restrict__ out_b,
        float* __restrict__ out) {
    __shared__ _Float16 feat[16 * FINP];   // 13 KB
    __shared__ _Float16 h1s[16 * A1P];     // 3 KB
    __shared__ float    h2s[16 * A2P];     // 3 KB

    const int lane = threadIdx.x;
    const int b0   = blockIdx.x * 16;

    for (int e = lane; e < 16 * A1P; e += 32) h1s[e] = (_Float16)0.f;

    // build all_feat = [dense(4) | other(128) | user_info(128) | item(128) | pad]
    for (int e = lane; e < 16 * FINP; e += 32) {
        int r = e / FINP, c = e % FINP;
        int gb = b0 + r;
        float v = 0.f;
        if (c < 4) {
            v = dense_inputs[gb * 4 + c];
        } else if (c < 132) {
            int cc = c - 4;
            if (cc < Dn) {
                int sp = sparse_inputs[gb * 2 + 0];
                v = embed_other[(size_t)sp * Dn + cc];
            } else {
                int sp = sparse_inputs[gb * 2 + 1];
                v = embed_other[(size_t)OTHER_V * Dn + (size_t)sp * Dn + (cc - Dn)];
            }
        } else if (c < 260) {
            v = user_ws[(size_t)gb * DKn + (c - 132)];
        } else if (c < FINn) {
            v = (float)qitem_ws[(size_t)gb * DKn + (c - 260)];
        }
        feat[e] = (_Float16)v;
    }
    __syncthreads();

    // layer 1: [16x416] x [416x80]
    for (int nt = 0; nt < 5; ++nt) {
        v8f acc = {};
        for (int kt = 0; kt < 13; ++kt) {
            v16h a  = ldsA_frag(feat, FINP, kt, lane);
            v16h bf = glbB_frag(F1t, FINP, nt, kt, lane);
            acc = wmma16(a, bf, acc);
        }
        store_act_f16(h1s, A1P, nt, acc, ffn_b1, ffn_a1, lane, A1n);
    }
    __syncthreads();

    // layer 2: [16x96] x [96x40(48)]
    for (int nt = 0; nt < 3; ++nt) {
        v8f acc = {};
        for (int kt = 0; kt < 3; ++kt) {
            v16h a  = ldsA_frag(h1s, A1P, kt, lane);
            v16h bf = glbB_frag(F2t, A1P, nt, kt, lane);
            acc = wmma16(a, bf, acc);
        }
        store_act_f32(h2s, A2P, nt, acc, ffn_b2, ffn_a2, lane, A2n);
    }
    __syncthreads();

    // layer 3: 40 -> 1 + sigmoid
    if (lane < 16) {
        float x = out_b[0];
        for (int n = 0; n < A2n; ++n)
            x += h2s[lane * A2P + n] * out_W[n];
        out[b0 + lane] = 1.f / (1.f + __expf(-x));
    }
}

// ---------------------------------------------------------------------------
extern "C" void kernel_launch(void* const* d_in, const int* in_sizes, int n_in,
                              void* d_out, int out_size, void* d_ws, size_t ws_size,
                              hipStream_t stream) {
    const float* dense  = (const float*)d_in[0];
    const int*   sparse = (const int*)d_in[1];
    const int*   seq    = (const int*)d_in[2];
    const int*   item   = (const int*)d_in[3];
    const float* eo     = (const float*)d_in[4];
    const float* ei     = (const float*)d_in[5];
    const float* ec     = (const float*)d_in[6];
    const float* aW1    = (const float*)d_in[7];
    const float* ab1    = (const float*)d_in[8];
    const float* aa1    = (const float*)d_in[9];
    const float* aW2    = (const float*)d_in[10];
    const float* ab2    = (const float*)d_in[11];
    const float* aa2    = (const float*)d_in[12];
    const float* aW3    = (const float*)d_in[13];
    const float* ab3    = (const float*)d_in[14];
    const float* fW1    = (const float*)d_in[15];
    const float* fb1    = (const float*)d_in[16];
    const float* fa1    = (const float*)d_in[17];
    const float* fW2    = (const float*)d_in[18];
    const float* fb2    = (const float*)d_in[19];
    const float* fa2    = (const float*)d_in[20];
    const float* oW     = (const float*)d_in[21];
    const float* ob     = (const float*)d_in[22];
    float* out = (float*)d_out;

    // workspace layout (all 32B-aligned)
    char* w = (char*)d_ws;
    _Float16* W1t = (_Float16*)w; w += (size_t)A1n * 512 * 2;       // 81920
    _Float16* W2t = (_Float16*)w; w += (size_t)A2P * A1P * 2;       // 9216
    _Float16* F1t = (_Float16*)w; w += (size_t)A1n * FINP * 2;      // 66560
    _Float16* F2t = (_Float16*)w; w += (size_t)A2P * A1P * 2;       // 9216
    float*    user_ws = (float*)w; w += (size_t)Bn * DKn * 4;       // 1 MB
    _Float16* qitem   = (_Float16*)w;                               // 512 KB

    transpose_pad_f16<<<(A1n * 512 + 255) / 256, 256, 0, stream>>>(aW1, W1t, 512, A1n, 512, A1n);
    transpose_pad_f16<<<(A2P * A1P + 255) / 256, 256, 0, stream>>>(aW2, W2t, A1n, A2n, A1P, A2P);
    transpose_pad_f16<<<(A1n * FINP + 255) / 256, 256, 0, stream>>>(fW1, F1t, FINn, A1n, FINP, A1n);
    transpose_pad_f16<<<(A2P * A1P + 255) / 256, 256, 0, stream>>>(fW2, F2t, A1n, A2n, A1P, A2P);

    din_attention<<<Bn, 128, 0, stream>>>(seq, item, ei, ec, W1t, W2t,
                                          ab1, aa1, ab2, aa2, aW3, ab3,
                                          user_ws, qitem);

    din_ffn<<<Bn / 16, 32, 0, stream>>>(dense, sparse, eo, user_ws, qitem,
                                        F1t, F2t, fb1, fa1, fb2, fa2,
                                        oW, ob, out);
}
